// GlmImageVisionVQProjector_7430293422280
// MI455X (gfx1250) — compile-verified, matrix operand order checked by reading, not run on hardware
//
#include <hip/hip_runtime.h>

// ---------------------------------------------------------------------------
// MI455X (gfx1250, wave32) implementation.
// Heavy matmuls -> v_wmma_f32_16x16x32_bf16 (fp32 accumulate).
// Tile fills -> global_load_async_to_lds_b128 (ASYNCcnt DMA path, no VGPR
// staging), published with s_wait_asynccnt + workgroup barrier.
// ---------------------------------------------------------------------------

typedef __attribute__((ext_vector_type(16))) __bf16 v16bf;
typedef __attribute__((ext_vector_type(8)))  float  v8f;

union Frag { v16bf v; unsigned int u[8]; };

#define BM 128
#define BN 128
#define BK 32
#define LDT 40   // LDS pitch in bf16: 80B rows => every 16B chunk 16B-aligned
                 // (required by async b128 LDS writes); frag gathers at
                 // 20-bank stride are >=16-distinct-bank, <=2-way conflict.

__device__ __forceinline__ v8f zero8() {
  v8f z;
#pragma unroll
  for (int i = 0; i < 8; ++i) z[i] = 0.0f;
  return z;
}

// k index held by VGPR vv for this lane-half (ISA 7.12.2, 16-bit A/B 16x32)
__device__ __forceinline__ int kmap(int vv, int hf) {
  return ((vv & 3) << 1) + ((vv >> 2) << 4) + (hf << 3);
}

// 16B global -> LDS direct DMA (ASYNCcnt). LDS offset = low 32 bits of the
// generic pointer (flat->LDS truncation per ISA 10.2).
__device__ __forceinline__ void async_copy16(const __bf16* gsrc, __bf16* ldst) {
  unsigned int lds = (unsigned int)(size_t)(void*)ldst;
  unsigned long long ga = (unsigned long long)(size_t)(const void*)gsrc;
  asm volatile("global_load_async_to_lds_b128 %0, %1, off"
               :: "v"(lds), "v"(ga) : "memory");
}
__device__ __forceinline__ void async_wait0() {
  asm volatile("s_wait_asynccnt 0x0" ::: "memory");
}

// ------------------------- conversion / prep kernels -----------------------

__global__ __launch_bounds__(256) void k_cvt_bf16(const float* __restrict__ in,
                                                  __bf16* __restrict__ out,
                                                  size_t n) {
  size_t i = (size_t)blockIdx.x * 256 + threadIdx.x;
  if (i < n) out[i] = (__bf16)in[i];
}

// out[n*K + k] = in[k*N + n]   (row-major KxN fp32 -> row-major NxK bf16)
__global__ __launch_bounds__(256) void k_transpose_cvt(const float* __restrict__ in,
                                                       __bf16* __restrict__ out,
                                                       int K, int N) {
  size_t idx = (size_t)blockIdx.x * 256 + threadIdx.x;
  size_t tot = (size_t)K * N;
  if (idx >= tot) return;
  size_t k = idx / N;
  size_t n = idx - k * N;
  out[n * (size_t)K + k] = (__bf16)in[idx];
}

// per-row L2 normalize fp32 [R,D] -> bf16 [R,D]; one block per row
__global__ __launch_bounds__(256) void k_rownorm_cvt(const float* __restrict__ in,
                                                     __bf16* __restrict__ out,
                                                     int D) {
  size_t row = blockIdx.x;
  const float* r = in + row * (size_t)D;
  __bf16* o = out + row * (size_t)D;
  float ss = 0.0f;
  for (int i = threadIdx.x; i < D; i += 256) { float v = r[i]; ss += v * v; }
  __shared__ float red[256];
  red[threadIdx.x] = ss;
  __syncthreads();
  for (int st = 128; st > 0; st >>= 1) {
    if (threadIdx.x < st) red[threadIdx.x] += red[threadIdx.x + st];
    __syncthreads();
  }
  float tot = red[0];
  float rn = (tot > 0.0f) ? rsqrtf(tot) : 0.0f;
  for (int i = threadIdx.x; i < D; i += 256) o[i] = (__bf16)(r[i] * rn);
}

// ------------------------- generic / conv bf16 GEMM ------------------------
// C[M,N] = A[M,K] * BT[N,K]^T (+bias[N]) (+residual[M,N]), fp32 out.
// CONV: A is logical im2col of t[M, Cin] with 3x3 SAME taps, K = 9*Cin.

template <bool CONV>
__global__ __launch_bounds__(256) void gemm_bf16_kernel(
    const __bf16* __restrict__ A, const __bf16* __restrict__ BT,
    const float* __restrict__ bias, const float* __restrict__ residual,
    float* __restrict__ out, int M, int N, int K, int Cin, int H, int W) {
  __shared__ __bf16 sA[2][BM * LDT];
  __shared__ __bf16 sB[2][BN * LDT];

  const int tid  = threadIdx.x;
  const int lane = tid & 31, wid = tid >> 5;
  const int lr = lane & 15, hf = lane >> 4;
  const int bm = blockIdx.y * BM, bn = blockIdx.x * BN;
  const int wm = (wid >> 2) * 64, wn = (wid & 3) * 32;
  const int HW = H * W;

  v8f acc[4][2];
#pragma unroll
  for (int mi = 0; mi < 4; ++mi)
#pragma unroll
    for (int ni = 0; ni < 2; ++ni) acc[mi][ni] = zero8();

  auto loadB = [&](int buf, int k0) {
#pragma unroll
    for (int p = 0; p < 2; ++p) {
      int idx = p * 256 + tid;
      int row = idx >> 2, c = idx & 3;
      async_copy16(BT + (size_t)(bn + row) * K + k0 + c * 8,
                   &sB[buf][row * LDT + c * 8]);
    }
  };
  auto loadA = [&](int buf, int k0) {
#pragma unroll
    for (int p = 0; p < 2; ++p) {
      int idx = p * 256 + tid;
      int row = idx >> 2, c = idx & 3;
      if (!CONV) {
        async_copy16(A + (size_t)(bm + row) * K + k0 + c * 8,
                     &sA[buf][row * LDT + c * 8]);
      } else {
        int kg  = k0 + c * 8;          // 16B chunk never straddles a tap
        int tap = kg / Cin;
        int ci  = kg - tap * Cin;
        int pix = bm + row;
        int b   = pix / HW;
        int rem = pix - b * HW;
        int h   = rem / W;
        int w   = rem - h * W;
        int h2  = h + tap / 3 - 1;
        int w2  = w + tap % 3 - 1;
        if ((unsigned)h2 < (unsigned)H && (unsigned)w2 < (unsigned)W) {
          async_copy16(A + ((size_t)b * HW + (size_t)h2 * W + w2) * Cin + ci,
                       &sA[buf][row * LDT + c * 8]);
        } else {  // spatial zero padding
          unsigned int* d = (unsigned int*)&sA[buf][row * LDT + c * 8];
          d[0] = 0u; d[1] = 0u; d[2] = 0u; d[3] = 0u;
        }
      }
    }
  };

  loadA(0, 0);
  loadB(0, 0);
  async_wait0();
  __syncthreads();
  const int nk = K / BK;
  for (int kt = 0; kt < nk; ++kt) {
    int cur = kt & 1;
    if (kt + 1 < nk) { loadA(cur ^ 1, (kt + 1) * BK); loadB(cur ^ 1, (kt + 1) * BK); }

    Frag fa[4], fb[2];
#pragma unroll
    for (int mi = 0; mi < 4; ++mi) {
      const __bf16* base = &sA[cur][(wm + mi * 16 + lr) * LDT];
#pragma unroll
      for (int vv = 0; vv < 8; ++vv)
        fa[mi].u[vv] = *(const unsigned int*)(base + kmap(vv, hf));
    }
#pragma unroll
    for (int ni = 0; ni < 2; ++ni) {
      const __bf16* base = &sB[cur][(wn + ni * 16 + lr) * LDT];
#pragma unroll
      for (int vv = 0; vv < 8; ++vv)
        fb[ni].u[vv] = *(const unsigned int*)(base + kmap(vv, hf));
    }
#pragma unroll
    for (int mi = 0; mi < 4; ++mi)
#pragma unroll
      for (int ni = 0; ni < 2; ++ni)
        acc[mi][ni] = __builtin_amdgcn_wmma_f32_16x16x32_bf16(
            false, fa[mi].v, false, fb[ni].v, (short)0, acc[mi][ni], false, false);
    async_wait0();     // settle this wave's async fills before publishing
    __syncthreads();
  }

  // epilogue: C lane layout (ISA 7.12.2): M = vv + 8*hf, N = lr within a tile
#pragma unroll
  for (int mi = 0; mi < 4; ++mi)
#pragma unroll
    for (int ni = 0; ni < 2; ++ni) {
      int n_g = bn + wn + ni * 16 + lr;
      float bv = bias ? bias[n_g] : 0.0f;
#pragma unroll
      for (int vv = 0; vv < 8; ++vv) {
        int m_g = bm + wm + mi * 16 + vv + hf * 8;
        size_t o = (size_t)m_g * N + n_g;
        float val = acc[mi][ni][vv] + bv;
        if (residual) val += residual[o];
        out[o] = val;
      }
    }
}

// ---------------- codebook argmax (zf @ emb_n^T, running argmax) -----------

__global__ __launch_bounds__(256) void argmax_kernel(
    const __bf16* __restrict__ A,   // zf  [M, K] bf16 (rows L2-normalized)
    const __bf16* __restrict__ BT,  // emb_n [KC, K] bf16
    float* __restrict__ cand_v, int* __restrict__ cand_i,
    int M, int KC, int K, int nsplit) {
  __shared__ __bf16 sA[2][BM * LDT];
  __shared__ __bf16 sB[2][BN * LDT];
  __shared__ float redv[128][4];
  __shared__ int   redi[128][4];

  const int tid  = threadIdx.x;
  const int lane = tid & 31, wid = tid >> 5;
  const int lr = lane & 15, hf = lane >> 4;
  const int bm = blockIdx.y * BM;
  const int wm = (wid >> 2) * 64, wn = (wid & 3) * 32;

  const int seg   = KC / nsplit;
  const int code0 = blockIdx.x * seg;

  float best[4][8];
  int   besti[4][8];
#pragma unroll
  for (int mi = 0; mi < 4; ++mi)
#pragma unroll
    for (int vv = 0; vv < 8; ++vv) { best[mi][vv] = -3.4e38f; besti[mi][vv] = 0; }

  auto loadA = [&](int buf, int k0) {
#pragma unroll
    for (int p = 0; p < 2; ++p) {
      int idx = p * 256 + tid;
      int row = idx >> 2, c = idx & 3;
      async_copy16(A + (size_t)(bm + row) * K + k0 + c * 8,
                   &sA[buf][row * LDT + c * 8]);
    }
  };
  auto loadB = [&](int buf, int k0, int bn) {
#pragma unroll
    for (int p = 0; p < 2; ++p) {
      int idx = p * 256 + tid;
      int row = idx >> 2, c = idx & 3;
      async_copy16(BT + (size_t)(bn + row) * K + k0 + c * 8,
                   &sB[buf][row * LDT + c * 8]);
    }
  };

  for (int nc = 0; nc < seg; nc += BN) {
    const int bn = code0 + nc;
    v8f acc[4][2];
#pragma unroll
    for (int mi = 0; mi < 4; ++mi)
#pragma unroll
      for (int ni = 0; ni < 2; ++ni) acc[mi][ni] = zero8();

    loadA(0, 0);
    loadB(0, 0, bn);
    async_wait0();
    __syncthreads();
    const int nk = K / BK;
    for (int kt = 0; kt < nk; ++kt) {
      int cur = kt & 1;
      if (kt + 1 < nk) { loadA(cur ^ 1, (kt + 1) * BK); loadB(cur ^ 1, (kt + 1) * BK, bn); }
      Frag fa[4], fb[2];
#pragma unroll
      for (int mi = 0; mi < 4; ++mi) {
        const __bf16* base = &sA[cur][(wm + mi * 16 + lr) * LDT];
#pragma unroll
        for (int vv = 0; vv < 8; ++vv)
          fa[mi].u[vv] = *(const unsigned int*)(base + kmap(vv, hf));
      }
#pragma unroll
      for (int ni = 0; ni < 2; ++ni) {
        const __bf16* base = &sB[cur][(wn + ni * 16 + lr) * LDT];
#pragma unroll
        for (int vv = 0; vv < 8; ++vv)
          fb[ni].u[vv] = *(const unsigned int*)(base + kmap(vv, hf));
      }
#pragma unroll
      for (int mi = 0; mi < 4; ++mi)
#pragma unroll
        for (int ni = 0; ni < 2; ++ni)
          acc[mi][ni] = __builtin_amdgcn_wmma_f32_16x16x32_bf16(
              false, fa[mi].v, false, fb[ni].v, (short)0, acc[mi][ni], false, false);
      async_wait0();
      __syncthreads();
    }
    // fold this 128-code chunk into the running per-row argmax
#pragma unroll
    for (int mi = 0; mi < 4; ++mi)
#pragma unroll
      for (int ni = 0; ni < 2; ++ni) {
        int idx = bn + wn + ni * 16 + lr;
#pragma unroll
        for (int vv = 0; vv < 8; ++vv) {
          float v = acc[mi][ni][vv];
          if (v > best[mi][vv] || (v == best[mi][vv] && idx < besti[mi][vv])) {
            best[mi][vv] = v; besti[mi][vv] = idx;
          }
        }
      }
    __syncthreads();
  }

  // reduce across the 16 lanes sharing a row (n dimension), then across waves
#pragma unroll
  for (int mi = 0; mi < 4; ++mi)
#pragma unroll
    for (int vv = 0; vv < 8; ++vv) {
      float v = best[mi][vv];
      int ix = besti[mi][vv];
      for (int off = 1; off < 16; off <<= 1) {
        float ov = __shfl_xor(v, off, 32);
        int   oi = __shfl_xor(ix, off, 32);
        if (ov > v || (ov == v && oi < ix)) { v = ov; ix = oi; }
      }
      if (lr == 0) {
        int rloc = wm + mi * 16 + vv + hf * 8;
        redv[rloc][wid & 3] = v;
        redi[rloc][wid & 3] = ix;
      }
    }
  __syncthreads();
  if (tid < 128) {
    float v = redv[tid][0];
    int ix = redi[tid][0];
#pragma unroll
    for (int c = 1; c < 4; ++c) {
      float ov = redv[tid][c]; int oi = redi[tid][c];
      if (ov > v || (ov == v && oi < ix)) { v = ov; ix = oi; }
    }
    size_t m_g = (size_t)(bm + tid);
    cand_v[m_g * nsplit + blockIdx.x] = v;
    cand_i[m_g * nsplit + blockIdx.x] = ix;
  }
}

// per-row: pick best split candidate, gather z_q = emb_n[idx] (bf16 copy)
__global__ __launch_bounds__(256) void k_select_gather(
    const float* __restrict__ cand_v, const int* __restrict__ cand_i,
    const __bf16* __restrict__ embn, __bf16* __restrict__ zq, int D, int nsplit) {
  __shared__ int sidx;
  size_t row = blockIdx.x;
  if (threadIdx.x == 0) {
    float v = cand_v[row * nsplit];
    int ix = cand_i[row * nsplit];
    for (int c = 1; c < nsplit; ++c) {
      float ov = cand_v[row * nsplit + c];
      int oi = cand_i[row * nsplit + c];
      if (ov > v || (ov == v && oi < ix)) { v = ov; ix = oi; }
    }
    sidx = ix;
  }
  __syncthreads();
  const unsigned int* src = (const unsigned int*)(embn + (size_t)sidx * D);
  unsigned int* dst = (unsigned int*)(zq + row * (size_t)D);
  for (int i = threadIdx.x; i < D / 2; i += 256) dst[i] = src[i];
}

// ------------------- fused GroupNorm (32 groups) + exact GELU --------------
// one block per (batch, group); x fp32 [B, HW, C]; out bf16 same layout
__global__ __launch_bounds__(256) void gn_gelu_kernel(
    const float* __restrict__ x, const float* __restrict__ gamma,
    const float* __restrict__ beta, __bf16* __restrict__ out,
    int HW, int C, int groups) {
  const int cpg = C / groups;
  const int b = blockIdx.x / groups, g = blockIdx.x % groups;
  const float* xb = x + (size_t)b * HW * C + (size_t)g * cpg;
  __bf16* ob = out + (size_t)b * HW * C + (size_t)g * cpg;
  const int n = HW * cpg;

  float s = 0.0f, ss = 0.0f;
  for (int i = threadIdx.x; i < n; i += 256) {
    int p = i / cpg, c = i - p * cpg;
    float v = xb[(size_t)p * C + c];
    s += v; ss += v * v;
  }
  __shared__ float rs[256], rss[256];
  rs[threadIdx.x] = s; rss[threadIdx.x] = ss;
  __syncthreads();
  for (int st = 128; st > 0; st >>= 1) {
    if (threadIdx.x < st) { rs[threadIdx.x] += rs[threadIdx.x + st]; rss[threadIdx.x] += rss[threadIdx.x + st]; }
    __syncthreads();
  }
  float mean = rs[0] / n;
  float var  = fmaxf(rss[0] / n - mean * mean, 0.0f);
  float rstd = rsqrtf(var + 1e-5f);

  for (int i = threadIdx.x; i < n; i += 256) {
    int p = i / cpg, c = i - p * cpg;
    float v  = xb[(size_t)p * C + c];
    float xn = (v - mean) * rstd * gamma[g * cpg + c] + beta[g * cpg + c];
    float ge = 0.5f * xn * (1.0f + erff(xn * 0.70710678118654752f));
    ob[(size_t)p * C + c] = (__bf16)ge;
  }
}

// ---------------------------------------------------------------------------

extern "C" void kernel_launch(void* const* d_in, const int* in_sizes, int n_in,
                              void* d_out, int out_size, void* d_ws, size_t ws_size,
                              hipStream_t stream) {
  const float* x     = (const float*)d_in[0];
  // d_in[1], d_in[2] are h,w scalars on device; reference fixes h=w=32.
  const float* qc_w  = (const float*)d_in[3];
  const float* qc_b  = (const float*)d_in[4];
  const float* emb   = (const float*)d_in[5];
  const float* pqc_w = (const float*)d_in[6];
  const float* pqc_b = (const float*)d_in[7];

  const int D  = in_sizes[4];             // 2048
  const int C  = in_sizes[3] / D;         // 1536
  const int M  = in_sizes[0] / C;         // 8192 (= B*H*W)
  const int KC = in_sizes[5] / D;         // 16384
  const int H = 32, W = 32;
  const int GROUPS = 32;
  const int KCONV = 9 * C;                // 13824
  const int NSPLIT = 4;
  float* out = (float*)d_out;

  // bump allocator over d_ws
  char* wp = (char*)d_ws;
  auto alloc = [&](size_t bytes) -> void* {
    void* r = (void*)wp;
    wp += (bytes + 255) & ~(size_t)255;
    return r;
  };
  __bf16* xb      = (__bf16*)alloc((size_t)M * C * 2);
  __bf16* wT_qc   = (__bf16*)alloc((size_t)D * C * 2);
  float*  z       = (float*)alloc((size_t)M * D * 4);
  __bf16* zf      = (__bf16*)alloc((size_t)M * D * 2);
  __bf16* embn    = (__bf16*)alloc((size_t)KC * D * 2);
  float*  cand_v  = (float*)alloc((size_t)M * NSPLIT * 4);
  int*    cand_i  = (int*)alloc((size_t)M * NSPLIT * 4);
  __bf16* zq      = (__bf16*)alloc((size_t)M * D * 2);
  __bf16* wT_pqc  = (__bf16*)alloc((size_t)C * D * 2);
  float*  y0      = (float*)alloc((size_t)M * C * 4);
  float*  y1      = (float*)alloc((size_t)M * C * 4);
  float*  c1out   = (float*)alloc((size_t)M * C * 4);
  __bf16* t       = (__bf16*)alloc((size_t)M * C * 2);
  __bf16* wT_conv = (__bf16*)alloc((size_t)C * KCONV * 2);

  auto blocks1d = [](size_t n) { return (unsigned)((n + 255) / 256); };

  // 1) quant_conv: z = x @ qc_w + qc_b
  k_cvt_bf16<<<blocks1d((size_t)M * C), 256, 0, stream>>>(x, xb, (size_t)M * C);
  k_transpose_cvt<<<blocks1d((size_t)C * D), 256, 0, stream>>>(qc_w, wT_qc, C, D);
  gemm_bf16_kernel<false><<<dim3(D / BN, M / BM), 256, 0, stream>>>(
      xb, wT_qc, qc_b, nullptr, z, M, D, C, 0, H, W);

  // 2) quantize: normalize, argmax of zf . emb_n^T, gather z_q = emb_n[idx]
  k_rownorm_cvt<<<M, 256, 0, stream>>>(z, zf, D);
  k_rownorm_cvt<<<KC, 256, 0, stream>>>(emb, embn, D);
  argmax_kernel<<<dim3(NSPLIT, M / BM), 256, 0, stream>>>(
      zf, embn, cand_v, cand_i, M, KC, D, NSPLIT);
  k_select_gather<<<M, 256, 0, stream>>>(cand_v, cand_i, embn, zq, D, NSPLIT);

  // 3) post_quant_conv: y0 = z_q @ pqc_w + pqc_b
  k_transpose_cvt<<<blocks1d((size_t)D * C), 256, 0, stream>>>(pqc_w, wT_pqc, D, C);
  gemm_bf16_kernel<false><<<dim3(C / BN, M / BM), 256, 0, stream>>>(
      zq, wT_pqc, pqc_b, nullptr, y0, M, C, D, 0, H, W);

  // 4) two residual blocks: y = conv2(gelu(gn2(conv1(gelu(gn1(y)))))) + y
  const int B = M / (H * W);
  for (int blk = 0; blk < 2; ++blk) {
    const float* c1w = (const float*)d_in[8 + blk * 8 + 0];
    const float* c1b = (const float*)d_in[8 + blk * 8 + 1];
    const float* n1g = (const float*)d_in[8 + blk * 8 + 2];
    const float* n1b = (const float*)d_in[8 + blk * 8 + 3];
    const float* c2w = (const float*)d_in[8 + blk * 8 + 4];
    const float* c2b = (const float*)d_in[8 + blk * 8 + 5];
    const float* n2g = (const float*)d_in[8 + blk * 8 + 6];
    const float* n2b = (const float*)d_in[8 + blk * 8 + 7];
    const float* pIn  = (blk == 0) ? y0 : y1;
    float*       pOut = (blk == 0) ? y1 : out;

    gn_gelu_kernel<<<B * GROUPS, 256, 0, stream>>>(pIn, n1g, n1b, t, H * W, C, GROUPS);
    k_transpose_cvt<<<blocks1d((size_t)KCONV * C), 256, 0, stream>>>(c1w, wT_conv, KCONV, C);
    gemm_bf16_kernel<true><<<dim3(C / BN, M / BM), 256, 0, stream>>>(
        t, wT_conv, c1b, nullptr, c1out, M, C, KCONV, C, H, W);

    gn_gelu_kernel<<<B * GROUPS, 256, 0, stream>>>(c1out, n2g, n2b, t, H * W, C, GROUPS);
    k_transpose_cvt<<<blocks1d((size_t)KCONV * C), 256, 0, stream>>>(c2w, wT_conv, KCONV, C);
    gemm_bf16_kernel<true><<<dim3(C / BN, M / BM), 256, 0, stream>>>(
        t, wT_conv, c2b, pIn, pOut, M, C, KCONV, C, H, W);
  }
}
